// GLCN_53240414601427
// MI455X (gfx1250) — compile-verified, matrix operand order checked by reading, not run on hardware
//
#include <hip/hip_runtime.h>

#define BB    64
#define NN    128
#define F_TOT 80
#define F_OBS 64

typedef __attribute__((ext_vector_type(2)))  float    v2f;
typedef __attribute__((ext_vector_type(8)))  float    v8f;
typedef __attribute__((ext_vector_type(16))) _Float16 v16h;

// Extract this lane's logit from the replicated-column C/D fragment,
// threshold it for A, and accumulate the log-likelihood term.
// 32-bit C/D 16x16 layout: vgpr e, lanes 0-15 -> M=e ; lanes 16-31 -> M=8+e.
// B columns were identical, so every lane holds every row's value.
__device__ __forceinline__ void glcn_epilogue(const v8f acc, int lane, int i0,
                                              int j, int b,
                                              float* __restrict__ Aout,
                                              float& psum) {
  if ((lane & 8) == 0) {              // active lanes: 0..7 and 16..23
    const int e = lane & 7;
    float L = acc[0];
    if (e == 1) L = acc[1];
    if (e == 2) L = acc[2];
    if (e == 3) L = acc[3];
    if (e == 4) L = acc[4];
    if (e == 5) L = acc[5];
    if (e == 6) L = acc[6];
    if (e == 7) L = acc[7];
    const int i = i0 + e + ((lane & 16) >> 1);   // rows 8..15 on lanes 16..23
    // y_hard = sigmoid(L) > 0.5  <=>  L > 0 ; diagonal forced to 1 afterwards
    const float aval = (i == j) ? 1.0f : (L > 0.0f ? 1.0f : 0.0f);
    Aout[((size_t)b * NN + i) * NN + j] = aval;
    if (i != j) {
      // selected = A*y + (1-A)*(1-y) with A=y_hard  ==  sigmoid(|L|)
      const float s = 1.0f / (1.0f + expf(-fabsf(L)));
      psum += logf(s + 1e-8f);
    }
  }
}

__global__ __launch_bounds__(256) void glcn_adj_kernel(
    const float* __restrict__ h, const float* __restrict__ a_link,
    float* __restrict__ Aout, float* __restrict__ partial) {
  __shared__ __align__(16) float sh[NN * F_OBS];   // 32 KB h_obs tile (f32)
  __shared__ float red[8];

  const int b    = blockIdx.x;       // batch
  const int tile = blockIdx.y;       // i-tile (16 rows)
  const int i0   = tile * 16;
  const int tid  = threadIdx.x;      // 256 threads = 8 wave32
  const int wave = tid >> 5;
  const int lane = tid & 31;

  // Stage h[b, :, 0:64] into LDS, dropping the trailing 16 unobserved feats.
  const float* hb = h + (size_t)b * NN * F_TOT;
  for (int t = tid; t < NN * (F_OBS / 4); t += 256) {   // 2048 float4 slots
    const int row = t >> 4;                              // 16 float4 per row
    const int k4  = (t & 15) << 2;
    const float4 v = *(const float4*)(hb + row * F_TOT + k4);
    *(float4*)(sh + row * F_OBS + k4) = v;
  }
  __syncthreads();

  const int g  = lane >> 4;              // lane group (K split)
  const int ri = i0 + (lane & 15);       // A-fragment row carried by this lane

  float psum = 0.0f;

#if __has_builtin(__builtin_amdgcn_wmma_f32_16x16x4_f32)
  // ---------- exact fp32 path: V_WMMA_F32_16X16X4_F32, K in chunks of 4 ----
  // A 16x4 f32 layout: vgpr r, group g -> K = r + 2g (rows = lane%16)
  // B  4x16 f32 layout (mirror): vgpr r, group g -> K = r + 2g (cols = lane%16)
  float hi[32];          // row-i values for this lane: K = 4c + 2g + {0,1}
  v2f   bfr[16];         // a_link broadcast across all 16 B columns
#pragma unroll
  for (int c = 0; c < 16; ++c) {
    hi[2 * c + 0] = sh[ri * F_OBS + 4 * c + 2 * g + 0];
    hi[2 * c + 1] = sh[ri * F_OBS + 4 * c + 2 * g + 1];
    bfr[c][0] = a_link[4 * c + 2 * g + 0];
    bfr[c][1] = a_link[4 * c + 2 * g + 1];
  }
  for (int j = wave * 16, jend = j + 16; j < jend; ++j) {
    v8f acc = {0.f, 0.f, 0.f, 0.f, 0.f, 0.f, 0.f, 0.f};
#pragma unroll
    for (int c = 0; c < 16; ++c) {
      const float2 hj = *(const float2*)(sh + j * F_OBS + 4 * c + 2 * g);
      v2f av;
      av[0] = fabsf(hi[2 * c + 0] - hj.x);
      av[1] = fabsf(hi[2 * c + 1] - hj.y);
      acc = __builtin_amdgcn_wmma_f32_16x16x4_f32(false, av, false, bfr[c],
                                                  (short)0, acc, false, false);
    }
    glcn_epilogue(acc, lane, i0, j, b, Aout, psum);
  }
#else
  // ---------- f16 fallback: V_WMMA_F32_16X16X32_F16 (probe-confirmed) ------
  // A 16x32 f16: group g, half m -> K = 32c + 8g + m + (m>=8 ? 8 : 0)
  // B 32x16 f16 as packed here (not A's order): group g, half m -> K = 32c + 16g + m
  float hi32[2][16];
  v16h  bf[2];
#pragma unroll
  for (int c = 0; c < 2; ++c)
#pragma unroll
    for (int m = 0; m < 16; ++m) {
      const int ka = 32 * c + 8 * g + m + ((m & 8) ? 8 : 0);
      const int kb = 32 * c + 16 * g + m;
      hi32[c][m] = sh[ri * F_OBS + ka];
      bf[c][m]   = (_Float16)a_link[kb];
    }
  for (int j = wave * 16, jend = j + 16; j < jend; ++j) {
    v8f acc = {0.f, 0.f, 0.f, 0.f, 0.f, 0.f, 0.f, 0.f};
#pragma unroll
    for (int c = 0; c < 2; ++c) {
      v16h af;
#pragma unroll
      for (int m = 0; m < 16; ++m) {
        const int ka = 32 * c + 8 * g + m + ((m & 8) ? 8 : 0);
        af[m] = (_Float16)fabsf(hi32[c][m] - sh[j * F_OBS + ka]);
      }
      acc = __builtin_amdgcn_wmma_f32_16x16x32_f16(false, af, false, bf[c],
                                                   (short)0, acc, false, false);
    }
    glcn_epilogue(acc, lane, i0, j, b, Aout, psum);
  }
#endif

  // Deterministic reduction of the log-likelihood partial sum.
#pragma unroll
  for (int off = 16; off > 0; off >>= 1) psum += __shfl_xor(psum, off, 32);
  if (lane == 0) red[wave] = psum;
  __syncthreads();
  if (tid == 0) {
    float t = 0.0f;
    for (int w = 0; w < 8; ++w) t += red[w];
    partial[b * 8 + tile] = t;
  }
}

__global__ void glcn_probs_kernel(const float* __restrict__ partial,
                                  float* __restrict__ probs) {
  const int b = threadIdx.x;
  if (b < BB) {
    float s = 0.0f;
    for (int t = 0; t < 8; ++t) s += partial[b * 8 + t];  // fixed order
    probs[b] = s;
  }
}

extern "C" void kernel_launch(void* const* d_in, const int* in_sizes, int n_in,
                              void* d_out, int out_size, void* d_ws,
                              size_t ws_size, hipStream_t stream) {
  (void)in_sizes; (void)n_in; (void)out_size; (void)ws_size;
  const float* h      = (const float*)d_in[0];
  const float* a_link = (const float*)d_in[1];
  // d_in[2] = rollout (==1): no gumbel noise, matches this forward path.
  float* Aout    = (float*)d_out;                       // [B, N, N]
  float* probs   = (float*)d_out + (size_t)BB * NN * NN;  // [B]
  float* partial = (float*)d_ws;                        // [B, 8] partial sums

  dim3 grid(BB, 8);
  glcn_adj_kernel<<<grid, 256, 0, stream>>>(h, a_link, Aout, partial);
  glcn_probs_kernel<<<1, 64, 0, stream>>>(partial, probs);
}
